// CircleLoss_40089224741572
// MI455X (gfx1250) — matrix operand order repaired and checked by reading, not compiled, not run on hardware
//
#include <hip/hip_runtime.h>
#include <cstdint>

typedef __attribute__((ext_vector_type(16))) _Float16 v16h;
typedef __attribute__((ext_vector_type(8)))  float    v8f;

#define N_ROWS 8192
#define DIMS   128
#define GAMMA_F 64.0f
#define MARGIN_F 0.25f

constexpr int COL_SLICES      = 8;
constexpr int COLS_PER_SLICE  = N_ROWS / COL_SLICES;      // 1024
constexpr int TILES_PER_SLICE = COLS_PER_SLICE / 16;      // 64

constexpr float LOG2E   = 1.4426950408889634f;
// exp(vp) = exp2(CP1 + CP2*(s-1)^2),  vp = gamma*(m^2 - (s-1)^2)
constexpr float CP1 = GAMMA_F * MARGIN_F * MARGIN_F * LOG2E;   //  4*log2e
constexpr float CP2 = -GAMMA_F * LOG2E;                        // -64*log2e
// exp(vn) = exp2((CN*(s-m)) * relu(s-m))
constexpr float CN  = GAMMA_F * LOG2E;                         //  64*log2e

// ---------------------------------------------------------------------------
// Kernel 1: L2-normalize each row of x (f32 [8192,128]) and store as f16.
// ---------------------------------------------------------------------------
__global__ __launch_bounds__(256) void normalize_f16_kernel(
    const float* __restrict__ x, _Float16* __restrict__ xh) {
  const int lane = threadIdx.x & 31;
  const int row  = blockIdx.x * 8 + (threadIdx.x >> 5);
  const float4* xr = (const float4*)(x + (size_t)row * DIMS);
  float4 v = xr[lane];
  float ss = v.x * v.x + v.y * v.y + v.z * v.z + v.w * v.w;
  #pragma unroll
  for (int m = 16; m >= 1; m >>= 1) ss += __shfl_xor(ss, m, 32);
  const float r = rsqrtf(ss);
  union { _Float16 h[4]; uint2 u; } p;
  p.h[0] = (_Float16)(v.x * r);
  p.h[1] = (_Float16)(v.y * r);
  p.h[2] = (_Float16)(v.z * r);
  p.h[3] = (_Float16)(v.w * r);
  ((uint2*)(xh + (size_t)row * DIMS))[lane] = p.u;
}

// ---------------------------------------------------------------------------
// Kernel 2: fused sim = x̂ x̂ᵀ (WMMA f32_16x16x32_f16) + circle-loss epilogue,
// software-pipelined: tile t+1's WMMA chain issues before tile t's epilogue,
// so epilogue VALU co-executes in the WMMA shadow.
// ---------------------------------------------------------------------------
__global__ __launch_bounds__(256) void circle_main_kernel(
    const _Float16* __restrict__ xh, const int* __restrict__ targets,
    float* __restrict__ partP, float* __restrict__ partN) {
  const int lane    = threadIdx.x & 31;
  const int wid     = (int)((blockIdx.x * blockDim.x + threadIdx.x) >> 5);
  const int rt      = wid >> 3;      // row tile 0..511
  const int cs      = wid & 7;       // column slice 0..7
  const int rowBase = rt * 16;
  const int laneLo  = lane & 15;
  const int hi      = lane >> 4;     // 0 or 1

  // A tiles (16 rows x 128 K), loaded once (contiguous k = 16*hi + s).
  const _Float16* arow = xh + (size_t)(rowBase + laneLo) * DIMS;
  v16h A[4];
  #pragma unroll
  for (int c = 0; c < 4; ++c) {
    const int kb = 32 * c + 16 * hi;
    union { uint4 u[2]; v16h v; } t;
    t.u[0] = *(const uint4*)(arow + kb);
    t.u[1] = *(const uint4*)(arow + kb + 8);
    A[c] = t.v;
  }

  int rowLab[8];
  #pragma unroll
  for (int v = 0; v < 8; ++v) rowLab[v] = targets[rowBase + v + 8 * hi];

  float sumP[8], sumN[8];
  #pragma unroll
  for (int v = 0; v < 8; ++v) { sumP[v] = 0.0f; sumN[v] = 0.0f; }

  const int j0 = cs * COLS_PER_SLICE;

  // One 16x16 sim tile: 4 chained WMMAs over K=128.
  auto computeTile = [&](int j) -> v8f {
    const _Float16* brow = xh + (size_t)(j + laneLo) * DIMS + 16 * hi;
    v8f a = {0.f, 0.f, 0.f, 0.f, 0.f, 0.f, 0.f, 0.f};
    #pragma unroll
    for (int c = 0; c < 4; ++c) {
      v16h B = *(const v16h*)(brow + 32 * c);
      a = __builtin_amdgcn_wmma_f32_16x16x32_f16(
          false, A[c], false, B, (short)0, a, false, false);
    }
    return a;
  };

  // Branchless epilogue, ONE v_exp_f32 per sim element: since exactly one of
  // the pos/neg terms is consumed, select the exp2 argument by the mask,
  // evaluate a single exp2, and route the result with cndmask arithmetic.
  auto epilogue = [&](const v8f& acc, int colLab) {
    #pragma unroll
    for (int v = 0; v < 8; ++v) {
      const float s    = acc[v];
      const float u    = s - 1.0f;
      const float argP = fmaf(CP2, u * u, CP1);            // log2 exp(-logit_p)
      const float t3   = s - MARGIN_F;
      const float argN = (CN * t3) * fmaxf(t3, 0.0f);      // log2 exp(logit_n)
      const bool  eq   = (rowLab[v] == colLab);
      const float e    = __builtin_amdgcn_exp2f(eq ? argP : argN);
      const float zP   = eq ? e : 0.0f;
      sumP[v] += zP;
      sumN[v] += (e - zP);
    }
  };

  // Software pipeline over 64 column tiles.
  v8f acc    = computeTile(j0);
  int colLab = targets[j0 + laneLo];
  for (int t = 1; t < TILES_PER_SLICE; ++t) {
    const int j        = j0 + t * 16;
    const int colLabN  = targets[j + laneLo];
    const v8f accN     = computeTile(j);
    epilogue(acc, colLab);
    acc = accN;
    colLab = colLabN;
  }
  epilogue(acc, colLab);

  // Reduce across the 16 lanes sharing each row (masks < 16 stay in half).
  #pragma unroll
  for (int v = 0; v < 8; ++v) {
    #pragma unroll
    for (int m = 8; m >= 1; m >>= 1) {
      sumP[v] += __shfl_xor(sumP[v], m, 32);
      sumN[v] += __shfl_xor(sumN[v], m, 32);
    }
  }
  if (laneLo == 0) {
    #pragma unroll
    for (int v = 0; v < 8; ++v) {
      const int row = rowBase + v + 8 * hi;
      partP[(size_t)row * COL_SLICES + cs] = sumP[v];
      partN[(size_t)row * COL_SLICES + cs] = sumN[v];
    }
  }
}

// ---------------------------------------------------------------------------
// Kernel 3: merge slices, logsumexp -> softplus -> mean.
// ---------------------------------------------------------------------------
__global__ __launch_bounds__(256) void circle_final_kernel(
    const float* __restrict__ partP, const float* __restrict__ partN,
    float* __restrict__ out) {
  const int row = (int)(blockIdx.x * blockDim.x + threadIdx.x);
  float sp = 0.0f, sn = 0.0f;
  #pragma unroll
  for (int k = 0; k < COL_SLICES; ++k) {
    sp += partP[(size_t)row * COL_SLICES + k];
    sn += partN[(size_t)row * COL_SLICES + k];
  }
  const float z    = logf(sp) + logf(sn);                    // lp + ln
  float loss = fmaxf(z, 0.0f) + log1pf(expf(-fabsf(z)));     // softplus
  #pragma unroll
  for (int m = 16; m >= 1; m >>= 1) loss += __shfl_xor(loss, m, 32);
  if ((threadIdx.x & 31) == 0) atomicAdd(out, loss * (1.0f / N_ROWS));
}

// ---------------------------------------------------------------------------
extern "C" void kernel_launch(void* const* d_in, const int* in_sizes, int n_in,
                              void* d_out, int out_size, void* d_ws, size_t ws_size,
                              hipStream_t stream) {
  const float* x       = (const float*)d_in[0];
  const int*   targets = (const int*)d_in[1];

  _Float16* xh    = (_Float16*)d_ws;                               // 2 MB
  float*    partP = (float*)((char*)d_ws + (size_t)N_ROWS * DIMS * sizeof(_Float16));
  float*    partN = partP + (size_t)N_ROWS * COL_SLICES;           // 2x 256 KB

  normalize_f16_kernel<<<N_ROWS / 8, 256, 0, stream>>>(x, xh);

  const int nWaves  = (N_ROWS / 16) * COL_SLICES;   // 4096
  circle_main_kernel<<<nWaves / 8, 256, 0, stream>>>(xh, targets, partP, partN);

  (void)hipMemsetAsync(d_out, 0, sizeof(float), stream);
  circle_final_kernel<<<N_ROWS / 256, 256, 0, stream>>>(partP, partN, (float*)d_out);
}
